// MoireAttention_77292231459670
// MI455X (gfx1250) — matrix operand order repaired and checked by reading, not compiled, not run on hardware
//
#include <hip/hip_runtime.h>
#include <hip/hip_bf16.h>

#define BB 8
#define NN 1024
#define IND 256
#define OUTD 256
#define HH 8
#define HD 32
#define QKVD 768

typedef __attribute__((ext_vector_type(16))) __bf16 v16bf;
typedef __attribute__((ext_vector_type(8)))  float  v8f;
typedef __attribute__((ext_vector_type(8)))  unsigned short u16x8;
typedef __attribute__((ext_vector_type(4)))  int    v4i;

union BFrag { v16bf v; u16x8 h[2]; };

#if defined(__HIP_DEVICE_COMPILE__) && __has_builtin(__builtin_amdgcn_global_load_async_to_lds_b128)
#define HAVE_ASYNC 1
#else
#define HAVE_ASYNC 0
#endif

__device__ __forceinline__ unsigned short f2bf(float f) {
  union { float f; unsigned u; } a; a.f = f;
  unsigned u = a.u + 0x7FFFu + ((a.u >> 16) & 1u);   // round-to-nearest-even
  return (unsigned short)(u >> 16);
}

// 16-byte global -> LDS copy; async path uses the CDNA5 async pipe
// (GLOBAL_LOAD_ASYNC_TO_LDS_B128, tracked with ASYNCcnt), fallback is a
// plain vector round-trip through VGPRs.
__device__ __forceinline__ void copy16_g2l(const void* g, void* l) {
#if HAVE_ASYNC
  __builtin_amdgcn_global_load_async_to_lds_b128(
      (__attribute__((address_space(1))) v4i*)g,
      (__attribute__((address_space(3))) v4i*)l, 0, 0);
#else
  *(u16x8*)l = *(const u16x8*)g;
#endif
}

__device__ __forceinline__ void wait_async_copies() {
#if HAVE_ASYNC
#if __has_builtin(__builtin_amdgcn_s_wait_asynccnt)
  __builtin_amdgcn_s_wait_asynccnt(0);
#else
  asm volatile("s_wait_asynccnt 0" ::: "memory");
#endif
#endif
}

// ---------------------------------------------------------------------------
// Kernel 1: QKV projection.  out[g][c] = sum_k x[g][k]*W[k][c] + b[c]
// 64x64 tile per block (128 threads / 4 waves), bf16 WMMA, fp32 accumulate.
// Q scaled by 1/sqrt(32).  Q,K written bf16 [B,H,N,32]; V written bf16
// TRANSPOSED [B,H,32,N] so the attention kernel can stage V tiles with pure
// (async) 16B copies.
// ---------------------------------------------------------------------------
__global__ __launch_bounds__(128) void qkv_proj_kernel(
    const float* __restrict__ x, const float* __restrict__ W,
    const float* __restrict__ bias,
    unsigned short* __restrict__ Qb, unsigned short* __restrict__ Kb,
    unsigned short* __restrict__ Vt)
{
  __shared__ __align__(16) unsigned short As[64 * 40];   // [row][k], stride 40
  __shared__ __align__(16) unsigned short Bs[64 * 40];   // transposed: [col][k]

  const int tid    = threadIdx.x;
  const int lane   = tid & 31;
  const int w      = tid >> 5;
  const int laneHi = lane >> 4;
  const int ln     = lane & 15;
  const int rowBase = blockIdx.x * 64;
  const int colBase = blockIdx.y * 64;

  v8f zero = {};
  v8f acc[4];
#pragma unroll
  for (int i = 0; i < 4; ++i) acc[i] = zero;

  for (int kb = 0; kb < IND; kb += 32) {
    __syncthreads();
    // x tile [64 rows x 32 k] -> As (coalesced read, fp32->bf16)
#pragma unroll
    for (int it = 0; it < 16; ++it) {
      int e = it * 128 + tid;
      int r = e >> 5, k = e & 31;
      As[r * 40 + k] = f2bf(x[(size_t)(rowBase + r) * IND + kb + k]);
    }
    // W tile [32 k x 64 cols] -> Bs transposed [col][k]
#pragma unroll
    for (int it = 0; it < 16; ++it) {
      int e = it * 128 + tid;
      int k = e >> 6, n = e & 63;
      Bs[n * 40 + k] = f2bf(W[(size_t)(kb + k) * QKVD + colBase + n]);
    }
    __syncthreads();

    // A fragment: row = lane%16 (+wave offset), k = laneHi*16 + {0..15} (contiguous)
    BFrag a;
    {
      const unsigned short* p = &As[(w * 16 + ln) * 40 + laneHi * 16];
      a.h[0] = *(const u16x8*)p;
      a.h[1] = *(const u16x8*)(p + 8);
    }
#pragma unroll
    for (int nb = 0; nb < 4; ++nb) {
      // B fragment: col = lane%16, k = laneHi*16 + {0..15} (contiguous)
      BFrag bm;
      const unsigned short* p = &Bs[(nb * 16 + ln) * 40 + laneHi * 16];
      bm.h[0] = *(const u16x8*)p;
      bm.h[1] = *(const u16x8*)(p + 8);
      acc[nb] = __builtin_amdgcn_wmma_f32_16x16x32_bf16(
          false, a.v, false, bm.v, (short)0, acc[nb], false, false);
    }
  }

  // Epilogue: add bias, route to Q/K/Vt, scale Q by 1/sqrt(32)
#pragma unroll
  for (int nb = 0; nb < 4; ++nb) {
    int col  = colBase + nb * 16 + ln;
    float bv = bias[col];
    int part = col >> 8;          // 0=Q 1=K 2=V
    int wi   = col & 255;
    int hh   = wi >> 5;
    int d    = wi & 31;
    unsigned short* dst = (part == 0) ? Qb : (part == 1) ? Kb : Vt;
    float mul = (part == 0) ? 0.17677669529663687f : 1.0f;  // 1/sqrt(32)
#pragma unroll
    for (int r = 0; r < 8; ++r) {
      int m = r + laneHi * 8;
      int g = rowBase + w * 16 + m;
      int b = g >> 10, n = g & 1023;
      float v = (acc[nb][r] + bv) * mul;
      size_t idx;
      if (part == 2)
        idx = ((size_t)(b * HH + hh) * HD + d) * NN + n;   // V transposed [.,d,n]
      else
        idx = ((size_t)(b * HH + hh) * NN + n) * HD + d;   // Q/K [.,n,d]
      dst[idx] = f2bf(v);
    }
  }
}

// ---------------------------------------------------------------------------
// Kernel 2: fused moire attention (flash-style online softmax).
// One block per (b, h, 64-row tile); 4 waves x 16 rows; 32 keys per step.
// K / V(t) / adj tiles staged with async global->LDS b128 copies (ASYNCcnt).
// S = Q K^T (WMMA) -> moire/mask/diag in fp32 -> online softmax ->
// P (bf16, via per-wave LDS transpose) x V (WMMA) accumulated in fp32.
// ---------------------------------------------------------------------------
__global__ __launch_bounds__(128) void moire_attn_kernel(
    const unsigned short* __restrict__ Qb, const unsigned short* __restrict__ Kb,
    const unsigned short* __restrict__ Vt, const float* __restrict__ adj,
    const int* __restrict__ mask, const float* __restrict__ shifts,
    const float* __restrict__ widths, float* __restrict__ out)
{
  __shared__ __align__(16) unsigned short Ks[32 * 40];    // [key][dim]
  __shared__ __align__(16) unsigned short Vs[32 * 40];    // [dim][key]
  __shared__ __align__(16) float Adj[64 * 36];            // [row][key], stride 36
  __shared__ __align__(16) unsigned short Pb[4 * 16 * 40];// per-wave P transpose

  const int tid = threadIdx.x, lane = tid & 31, w = tid >> 5;
  const int laneHi = lane >> 4, ln = lane & 15;
  const int bh = blockIdx.y;
  const int b = bh >> 3, h = bh & 7;
  const int rowBase = blockIdx.x * 64;

  const float sh = shifts[h];
  const float wd = widths[h];
  const float inv2w2 = 1.0f / (2.0f * wd * wd);
  const float LOGMIN = -20.72326583694641f;               // log(1e-9)

  // Q fragment (A layout), loaded once: row = rowBase + w*16 + lane%16
  BFrag qf;
  {
    const unsigned short* p =
        Qb + ((size_t)bh * NN + rowBase + w * 16 + ln) * HD + laneHi * 16;
    qf.h[0] = *(const u16x8*)p;
    qf.h[1] = *(const u16x8*)(p + 8);
  }

  v8f zero = {};
  v8f o0 = zero, o1 = zero;
  float mrow[8], lrow[8];
  int mi[8];
#pragma unroll
  for (int r = 0; r < 8; ++r) {
    mrow[r] = -3.0e38f;
    lrow[r] = 0.0f;
    int i = rowBase + w * 16 + r + laneHi * 8;
    mi[r] = mask[b * NN + i];
  }

  for (int j = 0; j < NN; j += 32) {
    __syncthreads();
    // --- stage tiles with 16B async global->LDS copies ---
    {
      // K tile: 32 keys x 32 dims bf16 (one 16B chunk per thread)
      int key = tid >> 2, d8 = (tid & 3) * 8;
      copy16_g2l(Kb + ((size_t)bh * NN + j + key) * HD + d8, &Ks[key * 40 + d8]);
      // V tile from transposed layout: 32 dims x 32 keys bf16
      int d = tid >> 2, k8 = (tid & 3) * 8;
      copy16_g2l(Vt + ((size_t)bh * HD + d) * NN + j + k8, &Vs[d * 40 + k8]);
    }
    // adj tile [64 rows x 32 keys] fp32: 4 chunks of 16B per thread
#pragma unroll
    for (int it = 0; it < 4; ++it) {
      int c = it * 128 + tid;            // 512 chunks total
      int r = c >> 3, f4 = (c & 7) * 4;  // 8 chunks per 32-float row
      copy16_g2l(&adj[((size_t)(b * NN + rowBase + r)) * NN + j + f4],
                 &Adj[r * 36 + f4]);
    }
    if (j + 32 < NN)  // hint next adj tile toward the caches
      __builtin_prefetch(&adj[((size_t)(b * NN + rowBase + (tid >> 1))) * NN + j + 32], 0, 1);
    wait_async_copies();
    __syncthreads();

    int mc0 = mask[b * NN + j + ln];
    int mc1 = mask[b * NN + j + 16 + ln];

    // K fragments (B layout): col = key, k = dim (contiguous per lane-half)
    BFrag kf0, kf1;
    {
      const unsigned short* p = &Ks[ln * 40 + laneHi * 16];
      kf0.h[0] = *(const u16x8*)p; kf0.h[1] = *(const u16x8*)(p + 8);
      p = &Ks[(16 + ln) * 40 + laneHi * 16];
      kf1.h[0] = *(const u16x8*)p; kf1.h[1] = *(const u16x8*)(p + 8);
    }
    v8f s0 = __builtin_amdgcn_wmma_f32_16x16x32_bf16(false, qf.v, false, kf0.v, (short)0, zero, false, false);
    v8f s1 = __builtin_amdgcn_wmma_f32_16x16x32_bf16(false, qf.v, false, kf1.v, (short)0, zero, false, false);

    // moire log-term (log(clip(exp(g))) == max(g, log 1e-9)), diagonal, mask
    float t0[8], t1[8];
#pragma unroll
    for (int r = 0; r < 8; ++r) {
      int m = r + laneHi * 8;
      int arow = w * 16 + m;
      float a0 = Adj[arow * 36 + ln];
      float a1 = Adj[arow * 36 + 16 + ln];
      float d0 = a0 - sh, d1 = a1 - sh;
      float lm0 = fmaxf(-d0 * d0 * inv2w2, LOGMIN);
      float lm1 = fmaxf(-d1 * d1 * inv2w2, LOGMIN);
      float v0 = s0[r] * lm0;
      float v1 = s1[r] * lm1;
      int gi = rowBase + arow;
      if (gi == j + ln)      v0 += 0.1f;
      if (gi == j + 16 + ln) v1 += 0.1f;
      if (!(mi[r] && mc0)) v0 = -1000.0f;
      if (!(mi[r] && mc1)) v1 = -1000.0f;
      t0[r] = v0; t1[r] = v1;
    }

    // online softmax per row (16-lane group reductions) + P -> LDS (A layout)
#pragma unroll
    for (int r = 0; r < 8; ++r) {
      float vm = fmaxf(t0[r], t1[r]);
      vm = fmaxf(vm, __shfl_xor(vm, 1, 32));
      vm = fmaxf(vm, __shfl_xor(vm, 2, 32));
      vm = fmaxf(vm, __shfl_xor(vm, 4, 32));
      vm = fmaxf(vm, __shfl_xor(vm, 8, 32));
      float mnew = fmaxf(mrow[r], vm);
      float sc   = __expf(mrow[r] - mnew);
      mrow[r] = mnew;
      float p0 = __expf(t0[r] - mnew);
      float p1 = __expf(t1[r] - mnew);
      float rs = p0 + p1;
      rs += __shfl_xor(rs, 1, 32);
      rs += __shfl_xor(rs, 2, 32);
      rs += __shfl_xor(rs, 4, 32);
      rs += __shfl_xor(rs, 8, 32);
      lrow[r] = lrow[r] * sc + rs;
      o0[r] *= sc;
      o1[r] *= sc;
      int m = r + laneHi * 8;
      Pb[(w * 16 + m) * 40 + ln]      = f2bf(p0);
      Pb[(w * 16 + m) * 40 + 16 + ln] = f2bf(p1);
    }

    // P (A layout) x V (B layout) -> accumulate output
    BFrag pf, vf0, vf1;
    {
      const unsigned short* p = &Pb[(w * 16 + ln) * 40 + laneHi * 16];
      pf.h[0] = *(const u16x8*)p; pf.h[1] = *(const u16x8*)(p + 8);
      p = &Vs[ln * 40 + laneHi * 16];
      vf0.h[0] = *(const u16x8*)p; vf0.h[1] = *(const u16x8*)(p + 8);
      p = &Vs[(16 + ln) * 40 + laneHi * 16];
      vf1.h[0] = *(const u16x8*)p; vf1.h[1] = *(const u16x8*)(p + 8);
    }
    o0 = __builtin_amdgcn_wmma_f32_16x16x32_bf16(false, pf.v, false, vf0.v, (short)0, o0, false, false);
    o1 = __builtin_amdgcn_wmma_f32_16x16x32_bf16(false, pf.v, false, vf1.v, (short)0, o1, false, false);
  }

  // normalize and write [B,N,H*32] fp32
#pragma unroll
  for (int r = 0; r < 8; ++r) {
    int m = r + laneHi * 8;
    int i = rowBase + w * 16 + m;
    float inv = 1.0f / lrow[r];
    size_t base = (size_t)(b * NN + i) * OUTD + h * HD;
    out[base + ln]      = o0[r] * inv;
    out[base + 16 + ln] = o1[r] * inv;
  }
}

// ---------------------------------------------------------------------------
extern "C" void kernel_launch(void* const* d_in, const int* in_sizes, int n_in,
                              void* d_out, int out_size, void* d_ws, size_t ws_size,
                              hipStream_t stream) {
  (void)in_sizes; (void)n_in; (void)out_size; (void)ws_size;
  const float* x      = (const float*)d_in[0];
  const float* adj    = (const float*)d_in[1];
  const int*   mask   = (const int*)d_in[2];
  const float* Wqkv   = (const float*)d_in[3];
  const float* bqkv   = (const float*)d_in[4];
  const float* shifts = (const float*)d_in[5];
  const float* widths = (const float*)d_in[6];
  float* out = (float*)d_out;

  const size_t per = (size_t)BB * HH * NN * HD;   // elems per Q/K/V buffer
  unsigned short* Qb = (unsigned short*)d_ws;
  unsigned short* Kb = Qb + per;
  unsigned short* Vt = Kb + per;                  // total 12 MB bf16 scratch

  dim3 g1(BB * NN / 64, QKVD / 64);
  qkv_proj_kernel<<<g1, 128, 0, stream>>>(x, Wqkv, bqkv, Qb, Kb, Vt);

  dim3 g2(NN / 64, BB * HH);
  moire_attn_kernel<<<g2, 128, 0, stream>>>(Qb, Kb, Vt, adj, mask, shifts, widths, out);
}
